// RNNEncoder_88003879895516
// MI455X (gfx1250) — matrix operand order repaired and checked by reading, not compiled, not run on hardware
//
#include <hip/hip_runtime.h>
#include <hip/hip_bf16.h>

// ---------------- problem constants ----------------
#define T_STEPS 8192
#define D_IN    2048
#define H_DIM   2048
#define G3H     6144   // 3*H
#define E_OUT   512

typedef __attribute__((ext_vector_type(8)))  float   v8f;
typedef __attribute__((ext_vector_type(8)))  __bf16  v8bf;
typedef __attribute__((ext_vector_type(16))) __bf16  v16bf;

// ---------------- fp32 -> bf16 convert (grid-stride) ----------------
__global__ __launch_bounds__(256) void k_cvt_bf16(const float* __restrict__ in,
                                                  __bf16* __restrict__ out, size_t n) {
    size_t i = (size_t)blockIdx.x * blockDim.x + threadIdx.x;
    size_t stride = (size_t)gridDim.x * blockDim.x;
    for (; i < n; i += stride) out[i] = (__bf16)in[i];
}

// ------- transpose + convert: in (R x Ncols, fp32) -> out (Ncols x R, bf16) -------
__global__ __launch_bounds__(256) void k_transpose_bf16(const float* __restrict__ in,
                                                        __bf16* __restrict__ out,
                                                        int R, int Ncols) {
    size_t n = (size_t)R * Ncols;
    size_t i = (size_t)blockIdx.x * blockDim.x + threadIdx.x;
    size_t stride = (size_t)gridDim.x * blockDim.x;
    for (; i < n; i += stride) {
        size_t j = i / R;           // output row  = input col
        size_t k = i % R;           // output col  = input row
        out[i] = (__bf16)in[k * (size_t)Ncols + j];
    }
}

// ---------------- WMMA bf16 GEMM: G[T,3H] = Xbf[T,K] @ WiT[3H,K]^T ----------------
// Double-buffered LDS. Block tile 128(M) x 128(N), K-chunk 32, 256 threads =
// 8 waves (4x2); each wave owns a 32x64 strip: 2 A-frags x 4 B-frags ->
// 8x v_wmma_f32_16x16x32_bf16 per K-chunk, one s_barrier per K-chunk.
#define MTILE 128
#define NTILE 128
#define KC    32
#define LDSP  48   // padded row (bf16): 96B -> 16B-aligned fragments, bank spread

__device__ __forceinline__ void stage_regs(const __bf16* __restrict__ A,
                                           const __bf16* __restrict__ BT,
                                           int bm, int bn, int sr, int sc, int k0,
                                           v8bf r[4]) {
    r[0] = *(const v8bf*)&A [(size_t)(bm + sr)      * D_IN + k0 + sc];
    r[1] = *(const v8bf*)&A [(size_t)(bm + sr + 64) * D_IN + k0 + sc];
    r[2] = *(const v8bf*)&BT[(size_t)(bn + sr)      * D_IN + k0 + sc];
    r[3] = *(const v8bf*)&BT[(size_t)(bn + sr + 64) * D_IN + k0 + sc];
}

__device__ __forceinline__ void stage_lds(__bf16 (*As)[LDSP], __bf16 (*Bs)[LDSP],
                                          int sr, int sc, const v8bf r[4]) {
    *(v8bf*)&As[sr][sc]      = r[0];
    *(v8bf*)&As[sr + 64][sc] = r[1];
    *(v8bf*)&Bs[sr][sc]      = r[2];
    *(v8bf*)&Bs[sr + 64][sc] = r[3];
}

__device__ __forceinline__ void compute_chunk(const __bf16 (*As)[LDSP],
                                              const __bf16 (*Bs)[LDSP],
                                              int wm, int wn, int lrow, int lhalf,
                                              v8f acc[2][4]) {
    v16bf af[2];
    #pragma unroll
    for (int mt = 0; mt < 2; ++mt) {
        const int row = wm * 32 + mt * 16 + lrow;
        const int kb  = lhalf * 8;
        v8bf lo = *(const v8bf*)&As[row][kb];
        v8bf hi = *(const v8bf*)&As[row][kb + 16];
        #pragma unroll
        for (int i = 0; i < 8; ++i) { af[mt][i] = lo[i]; af[mt][i + 8] = hi[i]; }
    }
    v16bf bf[4];
    #pragma unroll
    for (int nt = 0; nt < 4; ++nt) {
        const int col = wn * 64 + nt * 16 + lrow;
        const int ks  = lhalf * 8;
        v8bf lo = *(const v8bf*)&Bs[col][ks];
        v8bf hi = *(const v8bf*)&Bs[col][ks + 16];
        #pragma unroll
        for (int i = 0; i < 8; ++i) { bf[nt][i] = lo[i]; bf[nt][i + 8] = hi[i]; }
    }
    #pragma unroll
    for (int mt = 0; mt < 2; ++mt)
        #pragma unroll
        for (int nt = 0; nt < 4; ++nt)
            acc[mt][nt] = __builtin_amdgcn_wmma_f32_16x16x32_bf16(
                false, af[mt], false, bf[nt], (short)0, acc[mt][nt], false, false);
}

__global__ __launch_bounds__(256) void k_gemm_xwi(const __bf16* __restrict__ A,
                                                  const __bf16* __restrict__ BT,
                                                  float* __restrict__ C) {
    __shared__ __bf16 As0[MTILE][LDSP], Bs0[NTILE][LDSP];   // buffer 0
    __shared__ __bf16 As1[MTILE][LDSP], Bs1[NTILE][LDSP];   // buffer 1

    const int tid   = threadIdx.x;
    const int lane  = tid & 31;
    const int wave  = tid >> 5;
    const int wm    = wave & 3;          // M strip: 32 rows
    const int wn    = wave >> 2;         // N half:  64 cols
    const int bm    = blockIdx.y * MTILE;
    const int bn    = blockIdx.x * NTILE;

    const int sr = tid >> 2;             // 0..63 (and +64)
    const int sc = (tid & 3) * 8;        // 0,8,16,24

    const int lrow  = lane & 15;
    const int lhalf = lane >> 4;

    v8f acc[2][4] = {};
    v8bf r[4];

    // prologue: chunk 0 -> buffer 0
    stage_regs(A, BT, bm, bn, sr, sc, 0, r);
    stage_lds(As0, Bs0, sr, sc, r);
    __syncthreads();

    // steady state: 2 chunks per iteration, statically addressed buffers
    for (int k0 = 0; k0 < D_IN; k0 += 2 * KC) {
        // fetch chunk k0+KC while computing from buffer 0
        stage_regs(A, BT, bm, bn, sr, sc, k0 + KC, r);
        compute_chunk(As0, Bs0, wm, wn, lrow, lhalf, acc);
        stage_lds(As1, Bs1, sr, sc, r);
        __syncthreads();

        // fetch chunk k0+2*KC (if any) while computing from buffer 1
        const bool more = (k0 + 2 * KC) < D_IN;
        if (more) stage_regs(A, BT, bm, bn, sr, sc, k0 + 2 * KC, r);
        compute_chunk(As1, Bs1, wm, wn, lrow, lhalf, acc);
        if (more) {
            stage_lds(As0, Bs0, sr, sc, r);
            __syncthreads();
        }
    }

    // C/D layout: VGPR v -> M = v + lhalf*8 ; lanes[0:15] -> N = lrow
    #pragma unroll
    for (int mt = 0; mt < 2; ++mt) {
        #pragma unroll
        for (int nt = 0; nt < 4; ++nt) {
            const int col   = bn + wn * 64 + nt * 16 + lrow;
            const int rbase = bm + wm * 32 + mt * 16 + lhalf * 8;
            #pragma unroll
            for (int v = 0; v < 8; ++v)
                C[(size_t)(rbase + v) * G3H + col] = acc[mt][nt][v];
        }
    }
}

// ---------------- grid-wide barrier (persistent scan kernel) ----------------
__device__ inline void grid_barrier(int* cnt, volatile int* gen, int nblk) {
    __syncthreads();
    if (threadIdx.x == 0) {
        __threadfence();
        int g = *gen;
        if (atomicAdd(cnt, 1) == nblk - 1) {
            atomicExch(cnt, 0);
            __threadfence();
            atomicAdd((int*)gen, 1);
        } else {
            while (*gen == g) { __builtin_amdgcn_s_sleep(2); }
        }
        __threadfence();
    }
    __syncthreads();
}

__device__ inline float wave_dot_bf16(const __bf16* __restrict__ wrow,
                                      const float* __restrict__ vec, int lane) {
    float acc = 0.f;
    #pragma unroll
    for (int i = 0; i < 8; ++i) {
        const int base = i * 256 + lane * 8;
        v8bf wv = *(const v8bf*)&wrow[base];
        const float4* hp = (const float4*)&vec[base];
        float4 h0 = hp[0], h1 = hp[1];
        acc += h0.x * (float)wv[0] + h0.y * (float)wv[1]
             + h0.z * (float)wv[2] + h0.w * (float)wv[3]
             + h1.x * (float)wv[4] + h1.y * (float)wv[5]
             + h1.z * (float)wv[6] + h1.w * (float)wv[7];
    }
    #pragma unroll
    for (int m = 16; m > 0; m >>= 1) acc += __shfl_xor(acc, m, 32);
    return acc;
}

// ---------------- sequential GRU scan (persistent, 2 barriers/step) ----------------
#define NBLK 64
__global__ __launch_bounds__(256) void k_gru_scan(const float* __restrict__ G,
                                                  const float* __restrict__ b,
                                                  const __bf16* __restrict__ whT,
                                                  float* __restrict__ h,
                                                  float* __restrict__ z,
                                                  float* __restrict__ rh,
                                                  int* bar_cnt, int* bar_gen) {
    const int tid   = threadIdx.x;
    const int lane  = tid & 31;
    const int gwave = blockIdx.x * 8 + (tid >> 5);
    const int nwaves = NBLK * 8;

    // init state
    for (int i = blockIdx.x * 256 + tid; i < H_DIM; i += NBLK * 256) {
        h[i] = 0.f; z[i] = 0.f; rh[i] = 0.f;
    }
    grid_barrier(bar_cnt, bar_gen, NBLK);

    for (int t = 0; t < T_STEPS; ++t) {
        const float* gt = G + (size_t)t * G3H;

        // ---- phase A: zr = sigmoid(gx + h@Wh_zr + b) ; stash z and r*h ----
        for (int j = gwave; j < 2 * H_DIM; j += nwaves) {
            const __bf16* wrow = whT + (size_t)j * H_DIM;
            __builtin_prefetch(wrow + H_DIM, 0, 0);
            float acc = wave_dot_bf16(wrow, h, lane);
            if (lane == 0) {
                float s = 1.f / (1.f + __expf(-(gt[j] + b[j] + acc)));
                if (j < H_DIM) z[j] = s;
                else           rh[j - H_DIM] = s * h[j - H_DIM];
            }
        }
        grid_barrier(bar_cnt, bar_gen, NBLK);

        // ---- phase B: a = tanh(gx_a + (r*h)@Wh_a + b_a); h = (1-z)h + z*a ----
        for (int j = gwave; j < H_DIM; j += nwaves) {
            const int col = 2 * H_DIM + j;
            const __bf16* wrow = whT + (size_t)col * H_DIM;
            __builtin_prefetch(wrow + H_DIM, 0, 0);
            float acc = wave_dot_bf16(wrow, rh, lane);
            if (lane == 0) {
                float a  = tanhf(gt[col] + b[col] + acc);
                float zz = z[j];
                h[j] = (1.f - zz) * h[j] + zz * a;
            }
        }
        grid_barrier(bar_cnt, bar_gen, NBLK);
    }
}

// ---------------- epilogue: out = h @ w_out + b_out ----------------
__global__ __launch_bounds__(256) void k_out_proj(const float* __restrict__ h,
                                                  const float* __restrict__ wout,
                                                  const float* __restrict__ bout,
                                                  float* __restrict__ out) {
    int e = blockIdx.x * blockDim.x + threadIdx.x;
    if (e >= E_OUT) return;
    float acc = 0.f;
    for (int k = 0; k < H_DIM; ++k) acc += h[k] * wout[(size_t)k * E_OUT + e];
    out[e] = acc + bout[e];
}

// ---------------- launch ----------------
extern "C" void kernel_launch(void* const* d_in, const int* in_sizes, int n_in,
                              void* d_out, int out_size, void* d_ws, size_t ws_size,
                              hipStream_t stream) {
    const float* X  = (const float*)d_in[0];   // (T, D_IN)
    const float* Wi = (const float*)d_in[1];   // (D_IN, 3H)
    const float* Wh = (const float*)d_in[2];   // (H, 3H)
    const float* b  = (const float*)d_in[3];   // (3H,)
    const float* Wo = (const float*)d_in[4];   // (H, E)
    const float* bo = (const float*)d_in[5];   // (E,)
    float* out = (float*)d_out;

    char* ws = (char*)d_ws;
    size_t off = 0;
    auto carve = [&](size_t bytes) {
        size_t p = off;
        off = (off + bytes + 255) & ~(size_t)255;
        return p;
    };
    int*    bar  = (int*)   (ws + carve(256));                      // [cnt, gen]
    float*  h    = (float*) (ws + carve((size_t)H_DIM * 4));
    float*  z    = (float*) (ws + carve((size_t)H_DIM * 4));
    float*  rh   = (float*) (ws + carve((size_t)H_DIM * 4));
    __bf16* Xbf  = (__bf16*)(ws + carve((size_t)T_STEPS * D_IN * 2));
    __bf16* WiT  = (__bf16*)(ws + carve((size_t)G3H * D_IN * 2));   // (3H, K)
    __bf16* WhT  = (__bf16*)(ws + carve((size_t)G3H * H_DIM * 2));  // (3H, H)
    float*  G    = (float*) (ws + carve((size_t)T_STEPS * G3H * 4));

    hipMemsetAsync(bar, 0, 256, stream);

    k_cvt_bf16<<<2048, 256, 0, stream>>>(X, Xbf, (size_t)T_STEPS * D_IN);
    k_transpose_bf16<<<2048, 256, 0, stream>>>(Wi, WiT, D_IN, G3H);
    k_transpose_bf16<<<2048, 256, 0, stream>>>(Wh, WhT, H_DIM, G3H);

    dim3 gg(G3H / NTILE, T_STEPS / MTILE);   // (48, 64)
    k_gemm_xwi<<<gg, 256, 0, stream>>>(Xbf, WiT, G);

    k_gru_scan<<<NBLK, 256, 0, stream>>>(G, b, WhT, h, z, rh, bar, bar + 1);

    k_out_proj<<<(E_OUT + 255) / 256, 256, 0, stream>>>(h, Wo, bo, out);
}
